// NeuralODE_52063593562983
// MI455X (gfx1250) — matrix operand order repaired and checked by reading, not compiled, not run on hardware
//
#include <hip/hip_runtime.h>

#define T_STEPS 200
#define D_DIM   64
#define H_DIM   256

typedef __attribute__((ext_vector_type(16))) _Float16 v16h;
typedef __attribute__((ext_vector_type(8)))  float    v8f;

union Frag { v16h h; uint4 q[2]; };
union Acc  { v8f  v; float f[8]; };

__device__ __forceinline__ unsigned short f2h(float f) {
    _Float16 h = (_Float16)f;
    return __builtin_bit_cast(unsigned short, h);
}

// B-matrix fragment layout (16-bit, 32x16 tile per wmma):
// lane = (k_local/16)*16 + n%16 ; VGPR v holds K pairs: K = kk*32 + half*16 + v*2 + e
__device__ __forceinline__ int bfrag_off(int ntilesN, int k, int n) {
    int kk   = k >> 5;
    int kl   = k & 31;
    int half = kl >> 4;
    int kr   = kl & 15;
    int v    = kr >> 1;
    int e    = kr & 1;
    int lane = half * 16 + (n & 15);
    int nt   = n >> 4;
    return ((kk * ntilesN + nt) * 32 + lane) * 16 + v * 2 + e;
}

// A-matrix fragment layout (16-bit, 16x32 tile per wmma):
// lane = (K%32)/16*16 + m ; VGPR v holds K pairs: K = kk*32 + half*16 + v*2 + e
__device__ __forceinline__ int afrag_off(int k, int m) {
    int kk    = k >> 5;
    int kl    = k & 31;
    int upper = kl >> 4;
    int rem   = kl & 15;
    int half  = rem >> 3;
    int r2    = rem & 7;
    int v     = half * 4 + (r2 >> 1);
    int e     = r2 & 1;
    int lane  = upper * 16 + m;
    return (kk * 32 + lane) * 16 + v * 2 + e;
}

// Pre-swizzle all weights (f32 row-major) into f16 WMMA B-fragment layout in d_ws.
// ws halfword offsets: W1 @0 (16384), W2 @16384 (65536), W3 @81920 (16384).
__global__ __launch_bounds__(256) void prep_weights(const float* __restrict__ W1,
                                                    const float* __restrict__ W2,
                                                    const float* __restrict__ W3,
                                                    unsigned short* __restrict__ ws) {
    int i = blockIdx.x * 256 + threadIdx.x;
    if (i < 16384) {                 // W1: 64 x 256
        int k = i >> 8, n = i & 255;
        ws[bfrag_off(16, k, n)] = f2h(W1[i]);
    } else if (i < 81920) {          // W2: 256 x 256
        int j = i - 16384;
        int k = j >> 8, n = j & 255;
        ws[16384 + bfrag_off(16, k, n)] = f2h(W2[j]);
    } else if (i < 98304) {          // W3: 256 x 64
        int j = i - 81920;
        int k = j >> 6, n = j & 63;
        ws[81920 + bfrag_off(4, k, n)] = f2h(W3[j]);
    }
}

// One workgroup = 16 batch rows, resident for all 199 Euler steps.
// 4 waves (wave32); wave w owns y columns [16w,16w+16) as an f32 C-fragment.
// Hybrid weight placement: W2 (32/48 wmmas) stays register-stationary (256 VGPRs/wave),
// W1/W3 B-fragments live in LDS (64 KB) -> ~340 live VGPRs, epilogue gets ILP back.
__global__ __launch_bounds__(128) void ode_kernel(const float* __restrict__ y0,
                                                  const float* __restrict__ t,
                                                  const float* __restrict__ b1,
                                                  const float* __restrict__ b2,
                                                  const float* __restrict__ b3,
                                                  const unsigned short* __restrict__ wfrag,
                                                  float* __restrict__ out) {
    // W1 fragments @0 (16384 hw), W3 fragments @16384 (16384 hw)
    __shared__ __align__(16) unsigned short sW13[32768];
    // A-fragment staging: yA tiles [0..1], h1A tiles [2..9], h2A tiles [10..17]
    __shared__ __align__(16) unsigned short sA[18 * 512];

    const int tid  = threadIdx.x;
    const int lane = tid & 31;
    const int wave = tid >> 5;
    const int row0 = blockIdx.x * 16;

    const int ncol    = lane & 15;        // C-fragment column within tile
    const int mrow    = 8 * (lane >> 4);  // C-fragment row base
    const int ybase_n = wave * 16 + ncol; // this wave's y-column

    // One-time copy of W1/W3 fragments into LDS (16B vectors)
    {
        const uint4* s1 = reinterpret_cast<const uint4*>(wfrag);
        const uint4* s3 = reinterpret_cast<const uint4*>(wfrag + 81920);
        uint4*       d  = reinterpret_cast<uint4*>(sW13);
        for (int i = tid; i < 2048; i += 128) d[i]        = s1[i];
        for (int i = tid; i < 2048; i += 128) d[2048 + i] = s3[i];
    }

    const float dt = t[1] - t[0];

    // Per-lane bias scalars (C-fragment column is uniform across the 8 acc regs)
    float bv1[4], bv2[4];
    #pragma unroll
    for (int q = 0; q < 4; ++q) {
        bv1[q] = b1[(wave * 4 + q) * 16 + ncol];
        bv2[q] = b2[(wave * 4 + q) * 16 + ncol];
    }
    const float bv3 = b3[wave * 16 + ncol];

    // One-time load of this wave's W2 B-fragments into registers (32 x 32B/lane)
    Frag w2f[8][4];
    #pragma unroll
    for (int kk = 0; kk < 8; ++kk)
        #pragma unroll
        for (int q = 0; q < 4; ++q) {
            const uint4* p = reinterpret_cast<const uint4*>(
                wfrag + 16384 + ((kk * 16 + wave * 4 + q) * 32 + lane) * 16);
            w2f[kk][q].q[0] = p[0]; w2f[kk][q].q[1] = p[1];
        }

    // f32 master state for 16x16 slice of y, plus t=0 output slice
    float yv[8];
    #pragma unroll
    for (int cv = 0; cv < 8; ++cv) {
        int m = cv + mrow;
        float val = y0[(size_t)(row0 + m) * D_DIM + ybase_n];
        yv[cv] = val;
        out[((size_t)(row0 + m) * T_STEPS) * D_DIM + ybase_n] = val;
    }

    for (int tt = 1; tt < T_STEPS; ++tt) {
        // (1) scatter y (f16) into A-fragment layout, tiles 0..1
        #pragma unroll
        for (int cv = 0; cv < 8; ++cv)
            sA[afrag_off(ybase_n, cv + mrow)] = f2h(yv[cv]);
        __syncthreads();   // also covers the one-time sW13 fill before first GEMM1

        // (2) h1 = relu(y @ W1 + b1) : K=64 (2 k-tiles), N=256 (4 n-tiles/wave)
        {
            Acc acc[4];
            #pragma unroll
            for (int q = 0; q < 4; ++q)
                #pragma unroll
                for (int i = 0; i < 8; ++i) acc[q].f[i] = bv1[q];
            #pragma unroll
            for (int kk = 0; kk < 2; ++kk) {
                Frag a;
                const uint4* ap = reinterpret_cast<const uint4*>(sA + (kk * 32 + lane) * 16);
                a.q[0] = ap[0]; a.q[1] = ap[1];
                #pragma unroll
                for (int q = 0; q < 4; ++q) {
                    int nt = wave * 4 + q;
                    Frag b;
                    const uint4* bp = reinterpret_cast<const uint4*>(sW13 + ((kk * 16 + nt) * 32 + lane) * 16);
                    b.q[0] = bp[0]; b.q[1] = bp[1];
                    acc[q].v = __builtin_amdgcn_wmma_f32_16x16x32_f16(
                        false, a.h, false, b.h, (short)0, acc[q].v, false, false);
                }
            }
            #pragma unroll
            for (int q = 0; q < 4; ++q) {
                int kcol = (wave * 4 + q) * 16 + ncol;
                #pragma unroll
                for (int cv = 0; cv < 8; ++cv)
                    sA[2 * 512 + afrag_off(kcol, cv + mrow)] = f2h(fmaxf(acc[q].f[cv], 0.0f));
            }
        }
        __syncthreads();

        // (3) h2 = relu(h1 @ W2 + b2) : K=256 (8 k-tiles), 4 n-tiles/wave, B in VGPRs
        {
            Acc acc[4];
            #pragma unroll
            for (int q = 0; q < 4; ++q)
                #pragma unroll
                for (int i = 0; i < 8; ++i) acc[q].f[i] = bv2[q];
            #pragma unroll
            for (int kk = 0; kk < 8; ++kk) {
                Frag a;
                const uint4* ap = reinterpret_cast<const uint4*>(sA + ((2 + kk) * 32 + lane) * 16);
                a.q[0] = ap[0]; a.q[1] = ap[1];
                #pragma unroll
                for (int q = 0; q < 4; ++q)
                    acc[q].v = __builtin_amdgcn_wmma_f32_16x16x32_f16(
                        false, a.h, false, w2f[kk][q].h, (short)0, acc[q].v, false, false);
            }
            #pragma unroll
            for (int q = 0; q < 4; ++q) {
                int kcol = (wave * 4 + q) * 16 + ncol;
                #pragma unroll
                for (int cv = 0; cv < 8; ++cv)
                    sA[10 * 512 + afrag_off(kcol, cv + mrow)] = f2h(fmaxf(acc[q].f[cv], 0.0f));
            }
        }
        __syncthreads();

        // (4) y += dt * (h2 @ W3 + b3) : K=256 (8 k-tiles), N=64 (1 n-tile/wave)
        {
            Acc acc;
            #pragma unroll
            for (int i = 0; i < 8; ++i) acc.f[i] = bv3;
            #pragma unroll
            for (int kk = 0; kk < 8; ++kk) {
                Frag a;
                const uint4* ap = reinterpret_cast<const uint4*>(sA + ((10 + kk) * 32 + lane) * 16);
                a.q[0] = ap[0]; a.q[1] = ap[1];
                Frag b;
                const uint4* bp = reinterpret_cast<const uint4*>(sW13 + 16384 + ((kk * 4 + wave) * 32 + lane) * 16);
                b.q[0] = bp[0]; b.q[1] = bp[1];
                acc.v = __builtin_amdgcn_wmma_f32_16x16x32_f16(
                    false, a.h, false, b.h, (short)0, acc.v, false, false);
            }
            #pragma unroll
            for (int cv = 0; cv < 8; ++cv) {
                yv[cv] += dt * acc.f[cv];
                out[((size_t)(row0 + cv + mrow) * T_STEPS + tt) * D_DIM + ybase_n] = yv[cv];
            }
        }
        // no barrier needed here: next write target (yA) is two barriers from its last reader
    }
}

extern "C" void kernel_launch(void* const* d_in, const int* in_sizes, int n_in,
                              void* d_out, int out_size, void* d_ws, size_t ws_size,
                              hipStream_t stream) {
    (void)in_sizes; (void)n_in; (void)out_size; (void)ws_size;
    const float* y0 = (const float*)d_in[0];
    const float* t  = (const float*)d_in[1];
    const float* W1 = (const float*)d_in[2];
    const float* b1 = (const float*)d_in[3];
    const float* W2 = (const float*)d_in[4];
    const float* b2 = (const float*)d_in[5];
    const float* W3 = (const float*)d_in[6];
    const float* b3 = (const float*)d_in[7];
    unsigned short* wf = (unsigned short*)d_ws;   // 192 KB of f16 weight fragments
    float* out = (float*)d_out;

    prep_weights<<<384, 256, 0, stream>>>(W1, W2, W3, wf);      // 98304 elements
    ode_kernel<<<256, 128, 0, stream>>>(y0, t, b1, b2, b3, wf, out); // 4096/16 row-tiles
}
